// EnhancedGAT_7387343749408
// MI455X (gfx1250) — matrix operand order repaired and checked by reading, not compiled, not run on hardware
//
#include <hip/hip_runtime.h>
#include <hip/hip_bf16.h>
#include <cstdint>

// ---------------- problem constants (match reference) ----------------
#define NN    50000          // nodes
#define NE    800000         // edges (before self loops)
#define NET   (NE + NN)      // edges + self loops
#define G     256            // graphs
#define L     1000           // sequence length
#define H     4              // heads
#define C     32             // channels per head
#define FDIM  78             // input feature dim
#define FPAD  96             // padded to multiple of 32 for WMMA K-loop
#define EDIM  128            // embedding dim
#define LOUT  993            // L - K + 1
#define FEAT  128            // H*C
#define XTF   31776          // C * LOUT

typedef __attribute__((ext_vector_type(16))) _Float16     v16h;
typedef __attribute__((ext_vector_type(8)))  _Float16     v8h;
typedef __attribute__((ext_vector_type(8)))  float        v8f;
typedef __attribute__((ext_vector_type(4)))  unsigned int v4u;

#define GRID1(n) dim3((unsigned)(((n) + 255) / 256))

// Async global->LDS 16B copy (CDNA5 GLOBAL_LOAD_ASYNC_TO_LDS_B128, ASYNCcnt).
#define USE_ASYNC_LDS 1
static __device__ __forceinline__ void cp_b128_to_lds(_Float16* lds_dst,
                                                      const _Float16* gsrc) {
#if USE_ASYNC_LDS
  unsigned lo = (unsigned)(unsigned long long)(void*)lds_dst;  // LDS byte offset
  asm volatile("global_load_async_to_lds_b128 %0, %1, off"
               :: "v"(lo), "v"(gsrc) : "memory");
#else
  *(v4u*)lds_dst = *(const v4u*)gsrc;
#endif
}
static __device__ __forceinline__ void wait_async_lds() {
#if USE_ASYNC_LDS
  asm volatile("s_wait_asynccnt 0x0" ::: "memory");
#endif
}

// float atomic max via ordered-int transform (works for all IEEE floats)
static __device__ __forceinline__ void atomicMaxF(float* addr, float v) {
  int iv = __float_as_int(v);
  if (iv >= 0) atomicMax((int*)addr, iv);
  else         atomicMin((unsigned int*)addr, (unsigned int)iv);
}

// ---------------- small utility kernels ----------------
__global__ void k_fill(float* p, int n, float v) {
  int i = blockIdx.x * blockDim.x + threadIdx.x;
  if (i < n) p[i] = v;
}

__global__ void k_pad_x(const float* __restrict__ x, float* __restrict__ xp) {
  int i = blockIdx.x * blockDim.x + threadIdx.x;
  if (i >= NN * FPAD) return;
  int n = i / FPAD, k = i % FPAD;
  xp[i] = (k < FDIM) ? x[n * FDIM + k] : 0.0f;
}

// B [K,N] row-major f32  ->  Bt [N][Kp] f16 (zero-padded K)
__global__ void k_bt(const float* __restrict__ B, _Float16* __restrict__ Bt,
                     int K, int N, int Kp) {
  int i = blockIdx.x * blockDim.x + threadIdx.x;
  if (i >= N * Kp) return;
  int n = i / Kp, k = i % Kp;
  Bt[i] = (k < K) ? (_Float16)B[(long long)k * N + n] : (_Float16)0.0f;
}

// conv_w [C=32][ED=128][K=8] -> Bt [c][q=k*128+ed] f16 (im2col weight layout)
__global__ void k_bt_conv(const float* __restrict__ w, _Float16* __restrict__ Bt) {
  int i = blockIdx.x * blockDim.x + threadIdx.x;
  if (i >= 32 * 1024) return;
  int c = i >> 10, q = i & 1023;
  int k = q >> 7, ed = q & 127;
  Bt[i] = (_Float16)w[(c * 128 + ed) * 8 + k];
}

// gather emb[target] as f16: exh[(g*L+l)*128 + ed]
__global__ void k_emb(const int* __restrict__ target, const float* __restrict__ emb,
                      _Float16* __restrict__ exh) {
  int i = blockIdx.x * blockDim.x + threadIdx.x;
  if (i >= G * L * EDIM) return;
  int gl = i >> 7, ed = i & 127;
  int t = target[gl];
  exh[i] = (_Float16)emb[t * EDIM + ed];
}

// ---------------- WMMA GEMM: C[M,N] = A[M,K] * Bt^T, f16 inputs, f32 acc ----
// NT    : n-tiles (16 cols each) per wave (A-fragment reuse)
// AF16  : A already f16 (else f32 converted on the fly)
// AMODE : 0 = row offset m*lda ; 1 = conv im2col row map (g*L+lo)*128
// CMODE : 0 = C[m*ldc + coff + n] ; 1 = conv store C[g*XTF + n*LOUT + lo]
// SB    : >0 => stage 16*NT rows of Bt (SB halves capacity) into LDS via
//         async copies; WMMA B fragments then come from LDS (ds_load_b128).
template<int NT, bool AF16, int AMODE, int CMODE, int SB>
__global__ __launch_bounds__(128) void k_gemm(
    const void* __restrict__ Ap, const _Float16* __restrict__ Bt,
    const float* __restrict__ bias, float* __restrict__ Cp,
    int M, int K, int lda, int Kp, int ldc, int coff, int relu) {
  __shared__ __align__(64) _Float16 Bs[SB > 0 ? SB : 1];
  int lane = threadIdx.x & 31;
  int wv   = threadIdx.x >> 5;
  int n0   = blockIdx.y * (16 * NT);

  if (SB > 0) {  // cooperative async stage of this block's B rows (before guard!)
    int nch = (16 * NT * Kp) >> 3;                 // 16-byte chunks
    const _Float16* src = Bt + (long long)n0 * Kp;
    for (int cc = threadIdx.x; cc < nch; cc += 128)
      cp_b128_to_lds(Bs + cc * 8, src + cc * 8);
    wait_async_lds();
    __syncthreads();
  }

  int mt     = blockIdx.x * 4 + wv;
  int mtiles = (M + 15) >> 4;
  if (mt >= mtiles) return;          // wave-uniform: EXEC stays all-ones for WMMA
  int ml = lane & 15;                // A: row-in-tile, B/C: col-in-tile
  int hi = lane >> 4;                // half-wave select

  long long a_off;
  {
    int m = mt * 16 + ml;
    if (AMODE == 0) a_off = (long long)m * lda;
    else { int g = m / LOUT, lo = m % LOUT; a_off = ((long long)g * L + lo) * EDIM; }
  }
  const _Float16* brow[NT];
#pragma unroll
  for (int t = 0; t < NT; t++)
    brow[t] = Bt + (long long)(n0 + t * 16 + ml) * Kp + hi * 16;

  v8f zero = {};
  v8f acc[NT];
#pragma unroll
  for (int t = 0; t < NT; t++) acc[t] = zero;

  for (int k0 = 0; k0 < K; k0 += 32) {
    v16h a;
    if (AF16) {
      const _Float16* ap = (const _Float16*)Ap + a_off + k0 + hi * 16;
      v8h alo = *(const v8h*)ap;
      v8h ahi = *(const v8h*)(ap + 8);
      a = __builtin_shufflevector(alo, ahi, 0,1,2,3,4,5,6,7,8,9,10,11,12,13,14,15);
    } else {
      const float* ap = (const float*)Ap + a_off + k0 + hi * 16;
#pragma unroll
      for (int j = 0; j < 8; j++) a[j]     = (_Float16)ap[j];
#pragma unroll
      for (int j = 0; j < 8; j++) a[8 + j] = (_Float16)ap[8 + j];
    }
#pragma unroll
    for (int t = 0; t < NT; t++) {
      v16h b;
      if (SB > 0) b = *(const v16h*)(Bs + (t * 16 + ml) * Kp + hi * 16 + k0);
      else        b = *(const v16h*)(brow[t] + k0);
      acc[t] = __builtin_amdgcn_wmma_f32_16x16x32_f16(
          false, a, false, b, (short)0, acc[t], false, false);
    }
  }

  int mrb = mt * 16 + hi * 8;
#pragma unroll
  for (int t = 0; t < NT; t++) {
    int nc = n0 + t * 16 + ml;
    float bv = bias ? bias[nc] : 0.0f;
#pragma unroll
    for (int i = 0; i < 8; i++) {
      int mr = mrb + i;
      float v = acc[t][i] + bv;
      if (relu) v = fmaxf(v, 0.0f);
      if (CMODE == 0) Cp[(long long)mr * ldc + coff + nc] = v;
      else { int g = mr / LOUT, lo = mr % LOUT;
             Cp[(long long)g * XTF + nc * LOUT + lo] = v; }
    }
  }
}

// ---------------- GAT attention kernels ----------------
__global__ void k_att(const float* __restrict__ hW, const float* __restrict__ a_s,
                      const float* __restrict__ a_d, float* __restrict__ asrc,
                      float* __restrict__ adst) {
  int i = blockIdx.x * blockDim.x + threadIdx.x;
  if (i >= NN * H) return;
  int n = i >> 2, hd = i & 3;
  const float* hp = hW + n * FEAT + hd * C;
  float s1 = 0.f, s2 = 0.f;
#pragma unroll
  for (int c = 0; c < C; c++) { float hv = hp[c]; s1 += hv * a_s[hd*C+c]; s2 += hv * a_d[hd*C+c]; }
  asrc[i] = s1; adst[i] = s2;
}

__global__ void k_edge_p1(const int* __restrict__ ei, const float* __restrict__ asrc,
                          const float* __restrict__ adst, float* __restrict__ e,
                          float* __restrict__ mmax) {
  int i = blockIdx.x * blockDim.x + threadIdx.x;
  if (i >= NET * H) return;
  int eid = i >> 2, hd = i & 3;
  int s, d;
  if (eid < NE) { s = ei[eid]; d = ei[NE + eid]; } else { s = d = eid - NE; }
  float v = asrc[s * H + hd] + adst[d * H + hd];
  v = (v > 0.f) ? v : 0.2f * v;                 // leaky_relu 0.2
  e[i] = v;
  atomicMaxF(mmax + d * H + hd, v);
}

__global__ void k_edge_p2(const int* __restrict__ ei, float* __restrict__ e,
                          const float* __restrict__ mmax, float* __restrict__ denom) {
  int i = blockIdx.x * blockDim.x + threadIdx.x;
  if (i >= NET * H) return;
  int eid = i >> 2, hd = i & 3;
  int d = (eid < NE) ? ei[NE + eid] : eid - NE;
  float ex = expf(e[i] - mmax[d * H + hd]);
  e[i] = ex;                                    // cache exp for pass 3
  atomicAdd(denom + d * H + hd, ex);
}

__global__ void k_edge_p3(const int* __restrict__ ei, const float* __restrict__ e,
                          const float* __restrict__ denom, const float* __restrict__ hW,
                          float* __restrict__ agg) {
  int i = blockIdx.x * blockDim.x + threadIdx.x;
  if (i >= NET * FEAT) return;
  int eid = i >> 7, f = i & 127, hd = f >> 5;
  int s, d;
  if (eid < NE) { s = ei[eid]; d = ei[NE + eid]; } else { s = d = eid - NE; }
  float alpha = e[eid * H + hd] / (denom[d * H + hd] + 1e-16f);
  atomicAdd(agg + d * FEAT + f, hW[s * FEAT + f] * alpha);
}

__global__ void k_epilogue(const float* __restrict__ agg, const float* __restrict__ bias,
                           float* __restrict__ hout) {
  int i = blockIdx.x * blockDim.x + threadIdx.x;
  if (i >= NN * FEAT) return;
  hout[i] = fmaxf(agg[i] + bias[i & 127], 0.0f);
}

__global__ void k_pool(const float* __restrict__ h, const int* __restrict__ batch,
                       float* __restrict__ pooled) {
  int i = blockIdx.x * blockDim.x + threadIdx.x;
  if (i >= NN * FEAT) return;
  int n = i >> 7, f = i & 127;
  atomicAdd(pooled + batch[n] * FEAT + f, h[i]);
}

__global__ void k_out(const float* __restrict__ h2, const float* __restrict__ w,
                      const float* __restrict__ b, float* __restrict__ out) {
  int g = threadIdx.x;
  if (g >= G) return;
  float s = b[0];
  for (int j = 0; j < 256; j++) s += h2[g * 256 + j] * w[j];
  out[g] = s;
}

// ---------------- orchestration ----------------
extern "C" void kernel_launch(void* const* d_in, const int* in_sizes, int n_in,
                              void* d_out, int out_size, void* d_ws, size_t ws_size,
                              hipStream_t stream) {
  (void)in_sizes; (void)n_in; (void)out_size; (void)ws_size;
  const float* x      = (const float*)d_in[0];
  const int*   ei     = (const int*)  d_in[1];
  const int*   batch  = (const int*)  d_in[2];
  const int*   target = (const int*)  d_in[3];
  const float* W1     = (const float*)d_in[4];
  const float* as1    = (const float*)d_in[5];
  const float* ad1    = (const float*)d_in[6];
  const float* b1     = (const float*)d_in[7];
  const float* Wss    = (const float*)d_in[8];
  const float* ass    = (const float*)d_in[9];
  const float* ads    = (const float*)d_in[10];
  const float* bss    = (const float*)d_in[11];
  const float* xd_w   = (const float*)d_in[12];
  const float* xd_b   = (const float*)d_in[13];
  const float* emb    = (const float*)d_in[14];
  const float* cw     = (const float*)d_in[15];
  const float* cb     = (const float*)d_in[16];
  const float* xt_w   = (const float*)d_in[17];
  const float* xt_b   = (const float*)d_in[18];
  const float* f1_w   = (const float*)d_in[19];
  const float* f1_b   = (const float*)d_in[20];
  const float* f2_w   = (const float*)d_in[21];
  const float* f2_b   = (const float*)d_in[22];
  const float* o_w    = (const float*)d_in[23];
  const float* o_b    = (const float*)d_in[24];
  float* out = (float*)d_out;

  char* base = (char*)d_ws;
  size_t off = 0;
  auto alloc = [&](size_t bytes) -> char* {
    off = (off + 255) & ~(size_t)255;
    char* p = base + off;
    off += bytes;
    return p;
  };
  // persistent scratch (~28 MB)
  _Float16* Bt1   = (_Float16*)alloc((size_t)128 * FPAD * 2);
  _Float16* BtW   = (_Float16*)alloc((size_t)4 * 128 * 128 * 2);
  _Float16* Bt_xd = (_Float16*)alloc((size_t)128 * 128 * 2);
  _Float16* Bt_cv = (_Float16*)alloc((size_t)32 * 1024 * 2);
  _Float16* Bt_xt = (_Float16*)alloc((size_t)128 * XTF * 2);
  _Float16* Bt_f1 = (_Float16*)alloc((size_t)1024 * 256 * 2);
  _Float16* Bt_f2 = (_Float16*)alloc((size_t)256 * 1024 * 2);
  float* pooled = (float*)alloc((size_t)G * FEAT * 4);
  float* xc     = (float*)alloc((size_t)G * 256 * 4);
  float* h1b    = (float*)alloc((size_t)G * 1024 * 4);
  float* h2b    = (float*)alloc((size_t)G * 256 * 4);
  float* asrc   = (float*)alloc((size_t)NN * H * 4);
  float* adst   = (float*)alloc((size_t)NN * H * 4);
  float* mmax   = (float*)alloc((size_t)NN * H * 4);
  float* denom  = (float*)alloc((size_t)NN * H * 4);
  float* ebuf   = (float*)alloc((size_t)NET * H * 4);
  // aliased union region: GNN phase buffers (~96 MB) vs sequence phase (~98 MB)
  off = (off + 255) & ~(size_t)255;
  size_t uoff = off;
  float* x_pad = (float*)(base + uoff);             // 50000*96 f32
  float* h_cur = x_pad + (size_t)NN * FPAD;         // 50000*128 f32
  float* hW    = h_cur + (size_t)NN * FEAT;         // 50000*128 f32
  float* agg   = hW    + (size_t)NN * FEAT;         // 50000*128 f32
  _Float16* exh = (_Float16*)(base + uoff);         // 256*1000*128 f16
  float* cv = (float*)(base + ((uoff + (size_t)G * L * EDIM * 2 + 255) & ~(size_t)255));

  // ---- weight prep (f16 transposed, padded) ----
  k_pad_x<<<GRID1(NN * FPAD), 256, 0, stream>>>(x, x_pad);
  k_bt<<<GRID1(128 * FPAD), 256, 0, stream>>>(W1, Bt1, FDIM, 128, FPAD);
  for (int l = 0; l < 4; l++)
    k_bt<<<GRID1(128 * 128), 256, 0, stream>>>(Wss + l * 128 * 128, BtW + l * 128 * 128, 128, 128, 128);
  k_bt<<<GRID1(128 * 128), 256, 0, stream>>>(xd_w, Bt_xd, 128, 128, 128);
  k_bt_conv<<<GRID1(32 * 1024), 256, 0, stream>>>(cw, Bt_cv);
  k_bt<<<GRID1(128 * XTF), 256, 0, stream>>>(xt_w, Bt_xt, XTF, 128, XTF);
  k_bt<<<GRID1(1024 * 256), 256, 0, stream>>>(f1_w, Bt_f1, 256, 1024, 256);
  k_bt<<<GRID1(256 * 1024), 256, 0, stream>>>(f2_w, Bt_f2, 1024, 256, 1024);

  // ---- 5 GAT layers ----
  for (int l = 0; l < 5; l++) {
    const float*    A   = (l == 0) ? x_pad : h_cur;
    int             K   = (l == 0) ? FPAD : FEAT;
    const _Float16* Bt  = (l == 0) ? Bt1 : (BtW + (l - 1) * 128 * 128);
    const float*    a_s = (l == 0) ? as1 : (ass + (l - 1) * H * C);
    const float*    a_d = (l == 0) ? ad1 : (ads + (l - 1) * H * C);
    const float*    bb  = (l == 0) ? b1  : (bss + (l - 1) * FEAT);

    // hW = A @ W  (M=50000, N=128): NT=8 -> A read once; B staged in LDS (async)
    k_gemm<8, false, 0, 0, 128 * 128><<<dim3((3125 + 3) / 4, 1), 128, 0, stream>>>(
        A, Bt, nullptr, hW, NN, K, K, K, FEAT, 0, 0);

    k_att<<<GRID1(NN * H), 256, 0, stream>>>(hW, a_s, a_d, asrc, adst);
    k_fill<<<GRID1(NN * H), 256, 0, stream>>>(mmax, NN * H, -1e30f);
    k_fill<<<GRID1(NN * H), 256, 0, stream>>>(denom, NN * H, 0.0f);
    k_fill<<<GRID1(NN * FEAT), 256, 0, stream>>>(agg, NN * FEAT, 0.0f);
    k_edge_p1<<<GRID1(NET * H), 256, 0, stream>>>(ei, asrc, adst, ebuf, mmax);
    k_edge_p2<<<GRID1(NET * H), 256, 0, stream>>>(ei, ebuf, mmax, denom);
    k_edge_p3<<<GRID1(NET * FEAT), 256, 0, stream>>>(ei, ebuf, denom, hW, agg);
    k_epilogue<<<GRID1(NN * FEAT), 256, 0, stream>>>(agg, bb, h_cur);
  }

  // ---- pooling + xd branch ----
  k_fill<<<GRID1(G * FEAT), 256, 0, stream>>>(pooled, G * FEAT, 0.0f);
  k_pool<<<GRID1(NN * FEAT), 256, 0, stream>>>(h_cur, batch, pooled);
  // xd = relu(pooled @ xd_w + xd_b) -> xc[:, 0:128]
  k_gemm<8, false, 0, 0, 128 * 128><<<dim3(4, 1), 128, 0, stream>>>(
      pooled, Bt_xd, xd_b, xc, G, FEAT, FEAT, FEAT, 256, 0, 1);

  // ---- sequence branch: embed -> conv(im2col GEMM) -> fc_xt ----
  k_emb<<<GRID1(G * L * EDIM), 256, 0, stream>>>(target, emb, exh);
  // cv[g][c][lo] = conv: M = 256*993 rows, N=32, K=1024 (implicit im2col, lda=128)
  // conv weights (64 KB) staged in LDS once per block via async copies
  k_gemm<2, true, 1, 1, 32 * 1024><<<dim3(15888 / 4, 1), 128, 0, stream>>>(
      exh, Bt_cv, cb, cv, G * LOUT, 1024, 0, 1024, 0, 0, 0);
  // xt = relu(cv_flat @ xt_w + xt_b) -> xc[:, 128:256]  (B too big for LDS)
  k_gemm<4, false, 0, 0, 0><<<dim3(4, 2), 128, 0, stream>>>(
      cv, Bt_xt, xt_b, xc, G, XTF, XTF, XTF, 256, 128, 1);

  // ---- MLP head ----
  k_gemm<4, false, 0, 0, 0><<<dim3(4, 16), 128, 0, stream>>>(
      xc, Bt_f1, f1_b, h1b, G, 256, 256, 256, 1024, 0, 1);
  k_gemm<4, false, 0, 0, 0><<<dim3(4, 4), 128, 0, stream>>>(
      h1b, Bt_f2, f2_b, h2b, G, 1024, 1024, 1024, 256, 0, 1);
  k_out<<<1, 256, 0, stream>>>(h2b, o_w, o_b, out);
}
